// SymplecticODEEncoder_47493748359259
// MI455X (gfx1250) — matrix operand order repaired and checked by reading, not compiled, not run on hardware
//
#include <hip/hip_runtime.h>
#include <hip/hip_bf16.h>
#include <math.h>

// ---------------------------------------------------------------------------
// SymplecticODEEncoder for MI455X (gfx1250, wave32, WMMA f32<=f16 16x16x32)
//
//   stage 1: U0[t][b][z] = x[:, T-1-t, :] @ W_ih0.T + (b_ih0 + b_hh0)   (WMMA)
//   stage 2: sequential 2-layer tanh-RNN scan over t (16 batch tiles,
//            one wave each; W_hh0 / W_ih1 / W_hh1 live as B-fragments)  (WMMA)
//   stage 3: f = grad(V) closed form; 3 RK4 steps x 4 f-evals on
//            261888x16 rows; 76 WMMAs per f-eval per 16-row tile.
//            Packed weights staged into LDS once per WG via the
//            Tensor Data Mover (tensor_load_to_lds + s_wait_tensorcnt).
// ---------------------------------------------------------------------------

typedef __attribute__((ext_vector_type(16))) _Float16 v16h;
typedef __attribute__((ext_vector_type(8)))  _Float16 v8h;
typedef __attribute__((ext_vector_type(8)))  float    v8f;
typedef __attribute__((ext_vector_type(4)))  unsigned int v4u;
typedef __attribute__((ext_vector_type(8)))  int v8i;
typedef __attribute__((ext_vector_type(4)))  int v4i;

#define Bsz 256
#define Tsz 1024
#define Isz 64
#define Zsz 16
#define Hsz 128

// packed B-fragment weight region (f16 elements, contiguous):
//   [0)      p0t : 8 frags   (K=16p32, N=128)  B = Wp0.T
//   [4096)   p1t : 32 frags  (K=128,   N=128)  B = Wp1.T
//   [20480)  p1  : 32 frags  (K=128,   N=128)  B = Wp1
//   [36864)  p0  : 4 frags   (K=128,   N=16)   B = Wp0
#define WB_ODE_ELEMS 38912  // f16 elems = 77824 bytes = 9728 x 8B

__device__ __forceinline__ v8f splat8(float v) {
  v8f r;
#pragma unroll
  for (int i = 0; i < 8; ++i) r[i] = v;
  return r;
}

__device__ __forceinline__ float fast_tanh(float x) {
  // f16 WMMA already bounds precision; use v_exp_f32 fast path.
  x = fminf(fmaxf(x, -15.0f), 15.0f);
  float e = __expf(2.0f * x);
  return __fdividef(e - 1.0f, e + 1.0f);
}

__device__ __forceinline__ v8f wmma_f16(v16h a, v16h b, v8f c) {
  // D = A(16x32 f16) * B(32x16 f16) + C(16x16 f32)
  return __builtin_amdgcn_wmma_f32_16x16x32_f16(
      /*neg_a=*/false, a, /*neg_b=*/false, b,
      /*c_mod=*/(short)0, c, /*reuse_a=*/false, /*reuse_b=*/false);
}

// A-fragment per ISA 7.12.2: lanes 0-15 -> M=lane, K in {k0..k0+7} u {k0+16..k0+23}
// with k0 = kt*32 + (lane>=16 ? 8 : 0). Two aligned b128 LDS loads per lane.
__device__ __forceinline__ v16h load_a_frag(const _Float16* stage, int ldk,
                                            int lane, int kt) {
  const int half = lane >> 4;
  const int m    = lane & 15;
  const int k0   = kt * 32 + half * 8;
  v8h lo = *(const v8h*)(stage + m * ldk + k0);
  v8h hi = *(const v8h*)(stage + m * ldk + k0 + 16);
  v16h a;
#pragma unroll
  for (int i = 0; i < 8; ++i) { a[i] = lo[i]; a[8 + i] = hi[i]; }
  return a;
}

// B-fragments pre-swizzled (global ws or LDS): frag f, lane l holds 16
// contiguous f16 (one 32-byte load) = column strip K=khalf*8..+7 and 16+khalf*8..+7 (within the 32-k tile).
__device__ __forceinline__ v16h load_b_frag(const _Float16* wb, int fragid,
                                            int lane) {
  return *(const v16h*)(wb + (size_t)fragid * 512 + lane * 16);
}

__device__ __forceinline__ void wave_lds_sync() {
  // LDS-only fence at wavefront scope: orders DS ops without constraining
  // global traffic; wave_barrier pins the scheduler.
  __builtin_amdgcn_fence(__ATOMIC_ACQ_REL, "wavefront", "local");
  __builtin_amdgcn_wave_barrier();
}

// ---------------------------------------------------------------------------
// Pack a weight matrix into WMMA B-fragment layout (f16, zero-padded K).
//   B[k][n] = tr ? W[n*ldW + k] : W[k*ldW + n]
// ---------------------------------------------------------------------------
__global__ void pack_b_kernel(const float* __restrict__ W,
                              _Float16* __restrict__ dst, int Kreal, int Nreal,
                              int KT, int NT, int ldW, int tr) {
  int idx = blockIdx.x * blockDim.x + threadIdx.x;
  int total = KT * NT * 512;
  if (idx >= total) return;
  int f = idx >> 9;
  int rem = idx & 511;
  int lane = rem >> 4;
  int e = rem & 15;
  int kt = f / NT, nt = f - kt * NT;
  int khalf = lane >> 4, nl = lane & 15;
  int k = kt * 32 + (e < 8 ? khalf * 8 + e : 16 + khalf * 8 + (e - 8));
  int n = nt * 16 + nl;
  float v = 0.0f;
  if (k < Kreal && n < Nreal) v = tr ? W[n * ldW + k] : W[k * ldW + n];
  dst[idx] = (_Float16)v;
}

// ---------------------------------------------------------------------------
// Stage 1: U0[t][b][z] = xr[:,t,:] @ W_ih0.T + (b_ih0+b_hh0)
// ---------------------------------------------------------------------------
__global__ __launch_bounds__(32) void u0_kernel(
    const float* __restrict__ x, const float* __restrict__ b_ih0,
    const float* __restrict__ b_hh0, const _Float16* __restrict__ wb_ih0,
    float* __restrict__ U0) {
  __shared__ __attribute__((aligned(32))) _Float16 xs[16 * 64];
  const int lane = threadIdx.x;
  const int tile = blockIdx.x;
  const int t    = tile >> 4;
  const int b0   = (tile & 15) * 16;
  const int trev = (Tsz - 1) - t;  // xr[:,t,:] = x[:,T-1-t,:]
  {
    const int m = lane >> 1, c0 = (lane & 1) * 32;
    const float* src = x + ((size_t)(b0 + m) * Tsz + trev) * Isz + c0;
#pragma unroll
    for (int j = 0; j < 32; ++j) xs[m * 64 + c0 + j] = (_Float16)src[j];
  }
  wave_lds_sync();
  const int half = lane >> 4, n = lane & 15;
  v16h a0 = load_a_frag(xs, 64, lane, 0);
  v16h a1 = load_a_frag(xs, 64, lane, 1);
  v16h g0 = load_b_frag(wb_ih0, 0, lane);
  v16h g1 = load_b_frag(wb_ih0, 1, lane);
  v8f c = splat8(b_ih0[n] + b_hh0[n]);
  c = wmma_f16(a0, g0, c);
  c = wmma_f16(a1, g1, c);
#pragma unroll
  for (int i = 0; i < 8; ++i) {
    int m = i + half * 8;
    U0[((size_t)t * Bsz + b0 + m) * Zsz + n] = c[i];
  }
}

// ---------------------------------------------------------------------------
// Stage 2: sequential two-layer tanh RNN scan (16 blocks x 1 wave).
// ---------------------------------------------------------------------------
__global__ __launch_bounds__(32) void scan_kernel(
    const float* __restrict__ U0, const float* __restrict__ h0in,
    const float* __restrict__ b_ih1, const float* __restrict__ b_hh1,
    const _Float16* __restrict__ wb_hh0, const _Float16* __restrict__ wb_ih1,
    const _Float16* __restrict__ wb_hh1, float* __restrict__ rnn,
    float* __restrict__ out) {
  __shared__ __attribute__((aligned(32))) _Float16 sA[16 * 32];
  __shared__ __attribute__((aligned(32))) _Float16 sB[16 * 32];
  __shared__ __attribute__((aligned(32))) _Float16 sC[16 * 32];
  const int lane = threadIdx.x;
  const int half = lane >> 4, n = lane & 15;
  const int b0 = blockIdx.x * 16;
#pragma unroll
  for (int i = 0; i < 8; ++i) {  // zero K-pad columns 16..31 once
    int m = i + half * 8;
    sA[m * 32 + 16 + n] = (_Float16)0.0f;
    sB[m * 32 + 16 + n] = (_Float16)0.0f;
    sC[m * 32 + 16 + n] = (_Float16)0.0f;
  }
  v16h bf_hh0 = load_b_frag(wb_hh0, 0, lane);
  v16h bf_ih1 = load_b_frag(wb_ih1, 0, lane);
  v16h bf_hh1 = load_b_frag(wb_hh1, 0, lane);
  const float bias1 = b_ih1[n] + b_hh1[n];
  float h0c[8], h1c[8];
  const float h0v = h0in[n];
  const float h1v = h0in[Zsz + n];
#pragma unroll
  for (int i = 0; i < 8; ++i) { h0c[i] = h0v; h1c[i] = h1v; }

  for (int s = 0; s < Tsz; ++s) {
    wave_lds_sync();
#pragma unroll
    for (int i = 0; i < 8; ++i) {
      int m = i + half * 8;
      sA[m * 32 + n] = (_Float16)h0c[i];
      sB[m * 32 + n] = (_Float16)h1c[i];
    }
    wave_lds_sync();
    v16h aH0 = load_a_frag(sA, 32, lane, 0);
    v16h aH1 = load_a_frag(sB, 32, lane, 0);
    v8f c0;
#pragma unroll
    for (int i = 0; i < 8; ++i) {
      int m = i + half * 8;
      c0[i] = U0[((size_t)s * Bsz + b0 + m) * Zsz + n];
    }
    if (s + 1 < Tsz)
      __builtin_prefetch(&U0[((size_t)(s + 1) * Bsz + b0) * Zsz], 0, 1);
    c0 = wmma_f16(aH0, bf_hh0, c0);
#pragma unroll
    for (int i = 0; i < 8; ++i) h0c[i] = fast_tanh(c0[i]);
    wave_lds_sync();
#pragma unroll
    for (int i = 0; i < 8; ++i) {
      int m = i + half * 8;
      sC[m * 32 + n] = (_Float16)h0c[i];
    }
    wave_lds_sync();
    v16h aO0 = load_a_frag(sC, 32, lane, 0);
    v8f c1 = splat8(bias1);
    c1 = wmma_f16(aO0, bf_ih1, c1);
    c1 = wmma_f16(aH1, bf_hh1, c1);
#pragma unroll
    for (int i = 0; i < 8; ++i) h1c[i] = fast_tanh(c1[i]);
    const int torig = (Tsz - 1) - s;
#pragma unroll
    for (int i = 0; i < 8; ++i) {
      int m = i + half * 8;
      rnn[((size_t)(b0 + m) * Tsz + torig) * Zsz + n] = h1c[i];
    }
    if (s == 0) {  // ode_out[:, T-1, :] = rnn_out[:, T-1, :]
#pragma unroll
      for (int i = 0; i < 8; ++i) {
        int m = i + half * 8;
        out[((size_t)(b0 + m) * Tsz + (Tsz - 1)) * Zsz + n] = h1c[i];
      }
    }
  }
}

// ---------------------------------------------------------------------------
// Stage 3: RK4 over f = grad(V).  Weights TDM'd into LDS once per WG.
// ---------------------------------------------------------------------------
#define ODE_WAVES 4
#define NTILES (Bsz * (Tsz - 1) / 16)  // 16368
static_assert(NTILES % ODE_WAVES == 0, "grid must divide evenly");

__global__ __launch_bounds__(32 * ODE_WAVES) void ode_kernel(
    const float* __restrict__ rnn, const _Float16* __restrict__ wb_ode,
    const float* __restrict__ bp0, const float* __restrict__ bp1,
    const float* __restrict__ wfc, float* __restrict__ out) {
  __shared__ __attribute__((aligned(32))) _Float16 wlds[WB_ODE_ELEMS];
  __shared__ __attribute__((aligned(32))) _Float16 sh[ODE_WAVES][2][16 * 128];
  const int lane = threadIdx.x & 31;
  const int wave = threadIdx.x >> 5;
  const int tile = blockIdx.x * ODE_WAVES + wave;

  // ---- stage packed weights into LDS via the Tensor Data Mover -----------
#if __has_builtin(__builtin_amdgcn_tensor_load_to_lds) && \
    __has_builtin(__builtin_amdgcn_s_wait_tensorcnt)
  if (threadIdx.x < 32) {  // one wave issues the DMA (EXEC ignored by TDM)
    const unsigned E = WB_ODE_ELEMS * 2u / 8u;  // 9728 8-byte elements
    uint64_t ga = (uint64_t)(uintptr_t)wb_ode;
    uint32_t la = (uint32_t)(uintptr_t)&wlds[0];  // low 32 bits = LDS offset
    // D# group0: count=1 | lds_addr | global_addr[56:0] | type=2
    v4u g0 = {1u, la, (uint32_t)ga,
              ((uint32_t)(ga >> 32) & 0x01FFFFFFu) | 0x80000000u};
    // D# group1: data_size=3 (8B); tensor_dim0 = tile_dim0 = stride = E;
    //            tensor_dim1 = tile_dim1 = 1 (single row)
    v8i g1 = {(int)0x00030000,
              (int)((E & 0xFFFFu) << 16),          // tensor_dim0 lo16
              (int)((E >> 16) | (1u << 16)),       // tensor_dim0 hi | dim1 lo
              (int)((E & 0xFFFFu) << 16),          // dim1 hi=0 | tile_dim0=E
              (int)1,                              // tile_dim1 = 1
              (int)E,                              // tensor_dim0_stride lo32
              0, 0};
    v4i g2 = {0, 0, 0, 0};
    v4i g3 = {0, 0, 0, 0};
#if __clang_major__ >= 23
    v8i g4 = {0, 0, 0, 0, 0, 0, 0, 0};
    __builtin_amdgcn_tensor_load_to_lds(g0, g1, g2, g3, g4, 0);
#else
    __builtin_amdgcn_tensor_load_to_lds(g0, g1, g2, g3, 0);
#endif
    __builtin_amdgcn_s_wait_tensorcnt(0);
  }
#else
  // fallback: cooperative copy
  for (int i = threadIdx.x; i < WB_ODE_ELEMS / 8; i += 32 * ODE_WAVES)
    ((v8h*)wlds)[i] = ((const v8h*)wb_ode)[i];
#endif
  __syncthreads();

  const _Float16* Lp0t = wlds;           // 8 frags
  const _Float16* Lp1t = wlds + 4096;    // 32 frags
  const _Float16* Lp1  = wlds + 20480;   // 32 frags
  const _Float16* Lp0  = wlds + 36864;   // 4 frags
  _Float16* s1 = &sh[wave][0][0];
  _Float16* s2 = &sh[wave][1][0];
  const int half = lane >> 4, n = lane & 15;

  v16h fp0t[8];
#pragma unroll
  for (int nt = 0; nt < 8; ++nt) fp0t[nt] = load_b_frag(Lp0t, nt, lane);
  v16h fp0[4];
#pragma unroll
  for (int kt = 0; kt < 4; ++kt) fp0[kt] = load_b_frag(Lp0, kt, lane);
  float bp0v[8], bp1v[8], wfcv[8];
#pragma unroll
  for (int nt = 0; nt < 8; ++nt) {
    bp0v[nt] = bp0[nt * 16 + n];
    bp1v[nt] = bp1[nt * 16 + n];
    wfcv[nt] = wfc[nt * 16 + n];
  }

  // z tile: row = b*(T-1)+tt  <->  rnn[b][tt+1][:]
  const int row0 = tile * 16;
  int bidx[8], tidx[8];
  v8f z;
#pragma unroll
  for (int i = 0; i < 8; ++i) {
    int row = row0 + i + half * 8;
    int b = row / (Tsz - 1);
    int tt = row - b * (Tsz - 1);
    bidx[i] = b;
    tidx[i] = tt;
    z[i] = rnn[((size_t)b * Tsz + tt + 1) * Zsz + n];
  }

  auto fgrad = [&](v8f zin) -> v8f {
#pragma unroll
    for (int i = 0; i < 8; ++i) {  // stage z as A tile (K padded 16->32)
      int m = i + half * 8;
      s1[m * 128 + n] = (_Float16)zin[i];
      s1[m * 128 + 16 + n] = (_Float16)0.0f;
    }
    wave_lds_sync();
    v16h az = load_a_frag(s1, 128, lane, 0);
    wave_lds_sync();
    // forward p0: h1 = tanh(z@Wp0.T + bp0) -> s1 (16x128)
#pragma unroll
    for (int nt = 0; nt < 8; ++nt) {
      v8f c = splat8(bp0v[nt]);
      c = wmma_f16(az, fp0t[nt], c);
#pragma unroll
      for (int i = 0; i < 8; ++i) {
        int m = i + half * 8;
        s1[m * 128 + nt * 16 + n] = (_Float16)fast_tanh(c[i]);
      }
    }
    wave_lds_sync();
    v16h ah1[4];
#pragma unroll
    for (int kt = 0; kt < 4; ++kt) ah1[kt] = load_a_frag(s1, 128, lane, kt);
    // forward p1 + output grad: g_b = wfc*(1-tanh(b)^2) -> s2
#pragma unroll
    for (int nt = 0; nt < 8; ++nt) {
      v8f c = splat8(bp1v[nt]);
#pragma unroll
      for (int kt = 0; kt < 4; ++kt)
        c = wmma_f16(ah1[kt], load_b_frag(Lp1t, kt * 8 + nt, lane), c);
#pragma unroll
      for (int i = 0; i < 8; ++i) {
        int m = i + half * 8;
        float t = fast_tanh(c[i]);
        s2[m * 128 + nt * 16 + n] = (_Float16)(wfcv[nt] * (1.0f - t * t));
      }
    }
    wave_lds_sync();
    v16h agb[4];
#pragma unroll
    for (int kt = 0; kt < 4; ++kt) agb[kt] = load_a_frag(s2, 128, lane, kt);
    // backward p1: g_a = (g_b @ Wp1) * (1 - h1^2) -> s2
#pragma unroll
    for (int nt = 0; nt < 8; ++nt) {
      v8f c = splat8(0.0f);
#pragma unroll
      for (int kt = 0; kt < 4; ++kt)
        c = wmma_f16(agb[kt], load_b_frag(Lp1, kt * 8 + nt, lane), c);
#pragma unroll
      for (int i = 0; i < 8; ++i) {
        int m = i + half * 8;
        float h1 = (float)s1[m * 128 + nt * 16 + n];
        s2[m * 128 + nt * 16 + n] = (_Float16)(c[i] * (1.0f - h1 * h1));
      }
    }
    wave_lds_sync();
    v16h aga[4];
#pragma unroll
    for (int kt = 0; kt < 4; ++kt) aga[kt] = load_a_frag(s2, 128, lane, kt);
    // backward p0: f = g_a @ Wp0 (N=16)
    v8f g = splat8(0.0f);
#pragma unroll
    for (int kt = 0; kt < 4; ++kt) g = wmma_f16(aga[kt], fp0[kt], g);
    return g;
  };

  const float hs = 0.1f / 4.0f;  // DT / DISC
#pragma unroll 1
  for (int step = 0; step < 3; ++step) {  // DISC-1 RK4 steps
    v8f k1 = fgrad(z);
    v8f k2 = fgrad(z + (0.5f * hs) * k1);
    v8f k3 = fgrad(z + (0.5f * hs) * k2);
    v8f k4 = fgrad(z + hs * k3);
    z = z + (hs / 6.0f) * (k1 + 2.0f * k2 + 2.0f * k3 + k4);
  }
#pragma unroll
  for (int i = 0; i < 8; ++i)
    out[((size_t)bidx[i] * Tsz + tidx[i]) * Zsz + n] = z[i];
}

// ---------------------------------------------------------------------------
extern "C" void kernel_launch(void* const* d_in, const int* in_sizes, int n_in,
                              void* d_out, int out_size, void* d_ws,
                              size_t ws_size, hipStream_t stream) {
  const float* x     = (const float*)d_in[0];
  const float* h0    = (const float*)d_in[1];
  const float* W_ih0 = (const float*)d_in[2];
  const float* W_hh0 = (const float*)d_in[3];
  const float* b_ih0 = (const float*)d_in[4];
  const float* b_hh0 = (const float*)d_in[5];
  const float* W_ih1 = (const float*)d_in[6];
  const float* W_hh1 = (const float*)d_in[7];
  const float* b_ih1 = (const float*)d_in[8];
  const float* b_hh1 = (const float*)d_in[9];
  const float* Wp0   = (const float*)d_in[10];
  const float* bp0   = (const float*)d_in[11];
  const float* Wp1   = (const float*)d_in[12];
  const float* bp1   = (const float*)d_in[13];
  const float* Wfc   = (const float*)d_in[14];
  // bfc (d_in[15]) never appears in grad(V); seq_len (d_in[16]) fixed = 1024
  float* out = (float*)d_out;

  // workspace: U0 (16MB) | rnn_out (16MB) | packed f16 weights (~83KB)
  float* U0  = (float*)d_ws;
  float* rnn = U0 + (size_t)Tsz * Bsz * Zsz;
  _Float16* wb = (_Float16*)(rnn + (size_t)Tsz * Bsz * Zsz);
  _Float16* wb_ih0 = wb + 0;      // 2 frags  (K=64,  N=16)
  _Float16* wb_hh0 = wb + 1024;   // 1 frag   (K=16p, N=16)
  _Float16* wb_ih1 = wb + 1536;   // 1 frag
  _Float16* wb_hh1 = wb + 2048;   // 1 frag
  _Float16* wb_ode = wb + 2560;   // contiguous p0t|p1t|p1|p0 (38912 f16)
  _Float16* wb_p0t = wb_ode + 0;
  _Float16* wb_p1t = wb_ode + 4096;
  _Float16* wb_p1  = wb_ode + 20480;
  _Float16* wb_p0  = wb_ode + 36864;

  auto pack = [&](const float* W, _Float16* dst, int Kreal, int Nreal, int KT,
                  int NT, int ldW, int tr) {
    int total = KT * NT * 512;
    pack_b_kernel<<<(total + 127) / 128, 128, 0, stream>>>(W, dst, Kreal, Nreal,
                                                           KT, NT, ldW, tr);
  };
  pack(W_ih0, wb_ih0, 64, 16, 2, 1, 64, 1);
  pack(W_hh0, wb_hh0, 16, 16, 1, 1, 16, 1);
  pack(W_ih1, wb_ih1, 16, 16, 1, 1, 16, 1);
  pack(W_hh1, wb_hh1, 16, 16, 1, 1, 16, 1);
  pack(Wp0,   wb_p0t, 16, 128, 1, 8, 16, 1);
  pack(Wp1,   wb_p1t, 128, 128, 4, 8, 128, 1);
  pack(Wp1,   wb_p1,  128, 128, 4, 8, 128, 0);
  pack(Wp0,   wb_p0,  128, 16, 4, 1, 16, 0);

  u0_kernel<<<Tsz * (Bsz / 16), 32, 0, stream>>>(x, b_ih0, b_hh0, wb_ih0, U0);
  scan_kernel<<<Bsz / 16, 32, 0, stream>>>(U0, h0, b_ih1, b_hh1, wb_hh0,
                                           wb_ih1, wb_hh1, rnn, out);
  ode_kernel<<<NTILES / ODE_WAVES, 32 * ODE_WAVES, 0, stream>>>(
      rnn, wb_ode, bp0, bp1, Wfc, out);
}
